// FourierAttention_34437047779460
// MI455X (gfx1250) — matrix operand order repaired and checked by reading, not compiled, not run on hardware
//
#include <hip/hip_runtime.h>

#define DD   768
#define SS   256
#define BB   2
#define HH   12
#define WW   64
#define BS   (BB*SS)          // 512 rows in the QKV GEMM
#define CW   16               // K-chunk width (rows of K per double-buffer slot)
#define NCHUNK (SS/CW)        // 16 chunks
#define VPAD 8                // pad halves per V row to spread LDS banks

typedef __attribute__((ext_vector_type(2)))  float    v2f;
typedef __attribute__((ext_vector_type(8)))  float    v8f;
typedef __attribute__((ext_vector_type(8)))  _Float16 v8h;
typedef __attribute__((ext_vector_type(16))) _Float16 v16h;

// ---------------------------------------------------------------------------
// Kernel 1: QKV projection with fp32 WMMA (V_WMMA_F32_16X16X4_F32).
// out[m,n] = sum_d x[m,d] * W[n,d] + b[n]   (nn.Linear: x @ W.T + b)
// q,k stored f32 in (B,H,S,W); v stored f16 *transposed* in (B,H,W,S) so the
// attention kernel can bulk-async-copy it and read WMMA B-fragments contiguously.
// ---------------------------------------------------------------------------
__global__ __launch_bounds__(256) void qkv_wmma_kernel(
    const float* __restrict__ x,
    const float* __restrict__ Wq, const float* __restrict__ bq,
    const float* __restrict__ Wk, const float* __restrict__ bk,
    const float* __restrict__ Wv, const float* __restrict__ bv,
    float* __restrict__ ws)
{
  const int lane = threadIdx.x & 31;
  const int wave = threadIdx.x >> 5;
  int tile = blockIdx.x * 8 + wave;          // 4608 = 3 * (512/16) * (768/16)
  const int mat = tile / (32 * 48);
  tile %= (32 * 48);
  const int m0 = (tile / 48) * 16;
  const int n0 = (tile % 48) * 16;

  const float* Wmat = (mat == 0) ? Wq : (mat == 1) ? Wk : Wv;
  const float* bias = (mat == 0) ? bq : (mat == 1) ? bk : bv;

  // ISA A-layout 16x4 f32: lanes 0-15 hold K=0,1 ; lanes 16-31 hold K=2,3.
  // B-layout 4x16 mirrors: N = lane&15, same K split.
  const int lm = lane & 15;
  const int kk = (lane >> 4) * 2;

  const float* arow = x    + (size_t)(m0 + lm) * DD + kk;
  const float* brow = Wmat + (size_t)(n0 + lm) * DD + kk;

  v8f acc = {};
  for (int k0 = 0; k0 < DD; k0 += 4) {
    v2f a = *(const v2f*)(arow + k0);
    v2f b = *(const v2f*)(brow + k0);
    acc = __builtin_amdgcn_wmma_f32_16x16x4_f32(false, a, false, b,
                                                (short)0, acc, false, false);
  }

  const float bn = bias[n0 + lm];

  // C/D layout: VGPR r -> M=r (lanes 0-15) or M=r+8 (lanes 16-31), N=lane&15.
  const int n = n0 + lm;
  const int h = n / WW, w = n % WW;
  const int mBase = m0 + ((lane >> 4) << 3);

  if (mat < 2) {                      // q, k : f32 (B,H,S,W)
    float* o = ws + (size_t)mat * (BS * DD);
#pragma unroll
    for (int r = 0; r < 8; ++r) {
      int m = mBase + r;
      int b = m >> 8, s = m & 255;
      o[((size_t)(b * HH + h) * SS + s) * WW + w] = acc[r] + bn;
    }
  } else {                            // v : f16 transposed (B,H,W,S)
    _Float16* o = (_Float16*)(ws + (size_t)2 * BS * DD);
#pragma unroll
    for (int r = 0; r < 8; ++r) {
      int m = mBase + r;
      int b = m >> 8, s = m & 255;
      o[((size_t)(b * HH + h) * WW + w) * SS + s] = (_Float16)(acc[r] + bn);
    }
  }
}

// ---------------------------------------------------------------------------
// Kernel 2: fused sinc^4-kernel attention.
// Async-to-LDS staging (double-buffered K) overlapped with the transcendental
// score phase; softmax; P@V with f16 WMMA.
// ---------------------------------------------------------------------------
__global__ __launch_bounds__(256) void fourier_attn_kernel(
    const float* __restrict__ ws, const float* __restrict__ mask,
    float* __restrict__ out)
{
  __shared__ __align__(16) float    sQ[16][WW];          //  4 KB
  __shared__ __align__(16) _Float16 sVt[WW][SS + VPAD];  // 33 KB (V^T, padded)
  __shared__ __align__(16) float    sK[2][CW][WW];       //  8 KB double buffer
  __shared__ float                  sP[16][SS];          // 16 KB

  const int tid  = threadIdx.x;
  const int lane = tid & 31;
  const int wave = tid >> 5;

  const int blk    = blockIdx.x;
  const int rowBlk = blk & 15;
  const int bh     = blk >> 4;
  const int h      = bh % HH;
  const int b      = bh / HH;
  const int s0     = rowBlk * 16;

  const float*    qg = ws + ((size_t)(b * HH + h) * SS + s0) * WW;
  const float*    kg = ws + (size_t)BS * DD + (size_t)(b * HH + h) * SS * WW;
  const _Float16* vg = (const _Float16*)(ws + (size_t)2 * BS * DD)
                       + (size_t)(b * HH + h) * WW * SS;

  const unsigned ldsQ = (unsigned)(uintptr_t)&sQ[0][0];
  const unsigned ldsV = (unsigned)(uintptr_t)&sVt[0][0];
  const unsigned ldsK = (unsigned)(uintptr_t)&sK[0][0][0];

  // ---- async staging (ASYNCcnt path): Q tile, V^T panel, first K chunk -----
  {
    unsigned off = (unsigned)tid * 16u;                   // 4 KB Q
    asm volatile("global_load_async_to_lds_b128 %0, %1, %2"
                 :: "v"(ldsQ + off), "v"(off), "s"(qg) : "memory");
  }
#pragma unroll
  for (int i = 0; i < 8; ++i) {                           // 32 KB V^T, row-wise
    unsigned blkI = (unsigned)(tid + 256 * i);            // 16B block index
    unsigned row  = blkI >> 5, bi = blkI & 31;            // 32 blocks per row
    unsigned dst  = ldsV + row * ((SS + VPAD) * 2u) + bi * 16u;
    unsigned src  = blkI * 16u;
    asm volatile("global_load_async_to_lds_b128 %0, %1, %2"
                 :: "v"(dst), "v"(src), "s"(vg) : "memory");
  }
  {
    unsigned off = (unsigned)tid * 16u;                   // 4 KB K chunk 0
    asm volatile("global_load_async_to_lds_b128 %0, %1, %2"
                 :: "v"(ldsK + off), "v"(off), "s"(kg) : "memory");
  }

  // ---- Phase 1: scores, double-buffered K, overlapped with async loads -----
  const float PI_HALF = 1.57079632679489662f;  // pi * R, R = 0.5
  const int r   = tid >> 4;                    // one score entry per thread/chunk
  const int col = tid & 15;

  for (int c = 0; c < NCHUNK; ++c) {
    if (c + 1 < NCHUNK) {
      unsigned buf = (unsigned)((c + 1) & 1);
      unsigned off = (unsigned)tid * 16u;
      unsigned dst = ldsK + buf * (CW * WW * 4u) + off;
      unsigned src = (unsigned)((c + 1) * CW * WW * 4) + off;
      asm volatile("global_load_async_to_lds_b128 %0, %1, %2"
                   :: "v"(dst), "v"(src), "s"(kg) : "memory");
      asm volatile("s_wait_asynccnt 1" ::: "memory");  // older loads complete
    } else {
      asm volatile("s_wait_asynccnt 0" ::: "memory");
    }
    __syncthreads();

    const float* qr = sQ[r];
    const float* kr = sK[c & 1][col];
    float p = 1.0f;
#pragma unroll 16
    for (int wi = 0; wi < WW; ++wi) {
      int idx = (wi + lane) & 63;      // lane-rotated: conflict-free LDS banks
      float t = PI_HALF * (qr[idx] - kr[idx]);
      float s = (__builtin_fabsf(t) < 1e-6f) ? 1.0f : __sinf(t) / t;
      float s2 = s * s;
      p *= s2 * s2;
    }
    const int gcol = c * CW + col;
    sP[r][gcol] = p - 10000.0f * (1.0f - mask[b * SS + gcol]);
    __syncthreads();
  }

  // ---- Phase 2: row-wise softmax; each wave owns 2 rows, shfl reductions ---
  for (int rr = wave; rr < 16; rr += 8) {
    float m = -3.0e38f;
    for (int cc = lane; cc < SS; cc += 32) m = fmaxf(m, sP[rr][cc]);
#pragma unroll
    for (int off = 16; off > 0; off >>= 1) m = fmaxf(m, __shfl_xor(m, off, 32));
    float sum = 0.0f;
    for (int cc = lane; cc < SS; cc += 32) {
      float ev = __expf(sP[rr][cc] - m);
      sP[rr][cc] = ev;
      sum += ev;
    }
#pragma unroll
    for (int off = 16; off > 0; off >>= 1) sum += __shfl_xor(sum, off, 32);
    const float inv = 1.0f / sum;                // softmax + y/w.sum fold
    for (int cc = lane; cc < SS; cc += 32) sP[rr][cc] *= inv;
  }
  __syncthreads();

  // ---- Phase 3: y = P @ V with f16 WMMA; waves 0-3 own one 16x16 tile ------
  if (wave < 4) {
    const int n0t = wave * 16;
    v8f acc = {};
    for (int k0 = 0; k0 < SS; k0 += 32) {
      // A (16x32 f16): lanes 0-15 row=lane, K=k0+[0..7] then k0+16+[0..7];
      //                lanes 16-31 row=lane-16, K offsets shifted by +8.
      v16h a;
      const int rowA = lane & 15;
      const int kA   = k0 + ((lane >> 4) << 3);
#pragma unroll
      for (int i = 0; i < 8; ++i) a[i]     = (_Float16)sP[rowA][kA + i];
#pragma unroll
      for (int i = 0; i < 8; ++i) a[8 + i] = (_Float16)sP[rowA][kA + 16 + i];
      // B (32x16 f16): N=lane&15; same K order as A (lanes 16-31 shifted by +8).
      // V^T rows are contiguous in K -> two 16B LDS reads.
      const int nB = n0t + (lane & 15);
      const int kB = k0 + ((lane >> 4) << 3);
      v8h lo = *(const v8h*)&sVt[nB][kB];
      v8h hi = *(const v8h*)&sVt[nB][kB + 16];
      v16h bf = __builtin_shufflevector(lo, hi, 0, 1, 2, 3, 4, 5, 6, 7,
                                                8, 9, 10, 11, 12, 13, 14, 15);

      acc = __builtin_amdgcn_wmma_f32_16x16x32_f16(false, a, false, bf,
                                                   (short)0, acc, false, false);
    }
    // Store: out[b, s0+M, h*64 + n0t + N]  (merge-heads layout)
    const int N  = lane & 15;
    const int Mb = (lane >> 4) << 3;
#pragma unroll
    for (int rr = 0; rr < 8; ++rr) {
      const int srow = s0 + Mb + rr;
      out[((size_t)(b * SS + srow)) * DD + h * WW + n0t + N] = acc[rr];
    }
  }
}

// ---------------------------------------------------------------------------
extern "C" void kernel_launch(void* const* d_in, const int* in_sizes, int n_in,
                              void* d_out, int out_size, void* d_ws, size_t ws_size,
                              hipStream_t stream) {
  (void)in_sizes; (void)n_in; (void)out_size; (void)ws_size;
  const float* x    = (const float*)d_in[0];
  const float* mask = (const float*)d_in[1];
  const float* Wq   = (const float*)d_in[2];
  const float* bq   = (const float*)d_in[3];
  const float* Wk   = (const float*)d_in[4];
  const float* bk   = (const float*)d_in[5];
  const float* Wv   = (const float*)d_in[6];
  const float* bv   = (const float*)d_in[7];
  float* out = (float*)d_out;
  float* ws  = (float*)d_ws;   // q(f32) | k(f32) | v(f16, transposed (B,H,W,S))

  qkv_wmma_kernel<<<576, 256, 0, stream>>>(x, Wq, bq, Wk, bk, Wv, bv, ws);
  fourier_attn_kernel<<<BB * HH * (SS / 16), 256, 0, stream>>>(ws, mask, out);
}
